// LSTMAutoregressiveDecoder_56143812493672
// MI455X (gfx1250) — matrix operand order repaired and checked by reading, not compiled, not run on hardware
//
#include <hip/hip_runtime.h>

// ---------------------------------------------------------------------------
// LSTM autoregressive decoder for MI455X (gfx1250, wave32, WMMA bf16) — v3
// B=4096, U=256, O=128, S=48, gate width G=4U=1024
//
// Grid: 128 blocks (one per 32-row batch tile) x 256 threads (8 waves).
// Each wave owns u-slice [w*32, w*32+32) of hidden units, o-slice
// [w*16, w*16+16) of the dense output, and processes TWO 16-row M-tiles so
// every weight B-fragment (streamed from L2) feeds 2 WMMAs.
//
// v3: weight *offset* (not pointer) is laundered through inline asm each
// step — keeps global address-space inference (global_load_b128 with
// saddr+imm instead of flat_load), while still preventing the compiler from
// hoisting the 96 loop-invariant weight fragments out of the 48-step loop
// (which caused catastrophic VGPR spills in v1).
// ---------------------------------------------------------------------------

typedef __attribute__((ext_vector_type(16))) __bf16 v16bf;
typedef __attribute__((ext_vector_type(8)))  float  v8f;

constexpr int Bsz = 4096;
constexpr int U   = 256;
constexpr int O   = 128;
constexpr int S   = 48;
constexpr int G   = 1024;   // 4*U
constexpr int MT  = 2;      // M-tiles per wave
constexpr int ROWS = 16 * MT;

union FragCvt { uint4 u[2]; v16bf v; };

static __device__ __forceinline__ v16bf make_frag(uint4 lo, uint4 hi) {
  FragCvt f; f.u[0] = lo; f.u[1] = hi; return f.v;
}

// float -> bf16 bits, round-to-nearest-even
static __device__ __forceinline__ unsigned short f2bf(float f) {
  unsigned int u = __float_as_uint(f);
  u += 0x7FFFu + ((u >> 16) & 1u);
  return (unsigned short)(u >> 16);
}

// A fragment (16x32 bf16, row-major tile in LDS), ISA 16-bit A layout:
// lane m holds row m; dwords 0-3 = K[k0+8*half .. +7], dwords 4-7 = +16.
static __device__ __forceinline__ v16bf load_a_lds(const unsigned short* sh,
                                                   int ld, int lane, int k0) {
  const int m = lane & 15, half = lane >> 4;
  const unsigned short* p = sh + m * ld + k0 + 8 * half;
  uint4 lo = *(const uint4*)(p);
  uint4 hi = *(const uint4*)(p + 16);
  return make_frag(lo, hi);
}

// B fragment (32x16 bf16, row-major [K][N] weights in global/L2): lane L holds
// column N=n0+L%16; elements 0-7 = K[k0+8*half .. +7], elements 8-15 = +16.
static __device__ __forceinline__ v16bf load_b_g(const unsigned short* wgt,
                                                 int ld, int lane, int k0, int n0) {
  const unsigned short* p = wgt + (size_t)(k0 + 8 * (lane >> 4)) * ld + n0 + (lane & 15);
  union { unsigned short e[16]; v16bf v; } f;
#pragma unroll
  for (int s = 0; s < 16; ++s) f.e[s] = p[(size_t)(s < 8 ? s : s + 8) * ld];
  return f.v;
}

static __device__ __forceinline__ float fast_sigmoid(float x) {
  const float LOG2E = 1.4426950408889634f;
  float e = __builtin_amdgcn_exp2f(-x * LOG2E);
  return __builtin_amdgcn_rcpf(1.0f + e);
}

static __device__ __forceinline__ float fast_tanh(float x) {
  const float LOG2E = 1.4426950408889634f;
  float e = __builtin_amdgcn_exp2f(2.0f * x * LOG2E);
  return (e - 1.0f) * __builtin_amdgcn_rcpf(e + 1.0f);
}

// ---------------------------------------------------------------------------
// Prep: fp32 -> bf16 weight conversion into workspace
// ---------------------------------------------------------------------------
__global__ void cvt_bf16_kernel(const float* __restrict__ src,
                                unsigned short* __restrict__ dst, int n) {
  int i = blockIdx.x * blockDim.x + threadIdx.x;
  if (i < n) dst[i] = f2bf(src[i]);
}

// ---------------------------------------------------------------------------
// Persistent LSTM decoder kernel
// ---------------------------------------------------------------------------
__global__ __launch_bounds__(256)
void lstm_decoder_kernel(const float* __restrict__ x_in,   // [B,O]
                         const float* __restrict__ h0,     // [B,U]
                         const float* __restrict__ c0,     // [B,U]
                         const float* __restrict__ bias,   // [4U]
                         const float* __restrict__ dbias,  // [O]
                         const unsigned short* __restrict__ wk_g,  // [128][G] bf16
                         const unsigned short* __restrict__ wr_g,  // [256][G] bf16
                         const unsigned short* __restrict__ wd_g,  // [256][O] bf16
                         float* __restrict__ out)          // [B,S,O]
{
  __shared__ __align__(16) unsigned short sh_h[ROWS * U];  // 16 KB
  __shared__ __align__(16) unsigned short sh_x[ROWS * O];  // 8 KB

  const int tid  = threadIdx.x;
  const int lane = tid & 31;
  const int w    = tid >> 5;            // wave id 0..7
  const int b0   = blockIdx.x * ROWS;   // batch-tile base row
  const int col  = lane & 15;           // C/D frag: N column
  const int half = lane >> 4;           // C/D frag: row-half selector
  const int u0   = w * 32;              // this wave's hidden-unit slice
  const int o0   = w * 16;              // this wave's dense output slice

  // --- preload per-column biases (constant over rows & steps) ---
  float bsc[4][2];
#pragma unroll
  for (int g = 0; g < 4; ++g)
#pragma unroll
    for (int t = 0; t < 2; ++t)
      bsc[g][t] = bias[g * U + u0 + 16 * t + col];
  const float db = dbias[o0 + col];

  // --- cell state in registers, C-fragment layout, per M-tile ---
  float cfr[MT][2][8];
#pragma unroll
  for (int mt = 0; mt < MT; ++mt)
#pragma unroll
    for (int t = 0; t < 2; ++t)
#pragma unroll
      for (int r = 0; r < 8; ++r)
        cfr[mt][t][r] =
            c0[(size_t)(b0 + mt * 16 + r + 8 * half) * U + u0 + 16 * t + col];

  // --- stage h0 and last_input into LDS as bf16 ---
  for (int i = tid; i < ROWS * U; i += 256) {
    int rr = i / U, cc = i % U;
    sh_h[i] = f2bf(h0[(size_t)(b0 + rr) * U + cc]);
  }
  for (int i = tid; i < ROWS * O; i += 256) {
    int rr = i / O, cc = i % O;
    sh_x[i] = f2bf(x_in[(size_t)(b0 + rr) * O + cc]);
  }
  __syncthreads();

  for (int s = 0; s < S; ++s) {
    // Opaque zero offset: prevents hoisting/CSE of weight loads across the
    // step loop, but keeps the base pointers as kernel args so the loads
    // stay in the GLOBAL address space (global_load_b128, saddr+imm form,
    // LOADcnt only) instead of degrading to FLAT (LOADcnt+DScnt).
    unsigned int zoff = 0;
    asm volatile("" : "+s"(zoff));
    const unsigned short* wk = wk_g + zoff;
    const unsigned short* wr = wr_g + zoff;
    const unsigned short* wd = wd_g + zoff;

    // ============ Phase 1: gate GEMM  z = x@Wk + h@Wr + bias ============
    v8f acc[4][2][MT];
#pragma unroll
    for (int g = 0; g < 4; ++g)
#pragma unroll
      for (int t = 0; t < 2; ++t)
#pragma unroll
        for (int mt = 0; mt < MT; ++mt) {
          v8f a;
#pragma unroll
          for (int r = 0; r < 8; ++r) a[r] = bsc[g][t];
          acc[g][t][mt] = a;
        }

    // x part: K = 128
#pragma unroll
    for (int kc = 0; kc < O; kc += 32) {
      v16bf a0 = load_a_lds(sh_x,          O, lane, kc);
      v16bf a1 = load_a_lds(sh_x + 16 * O, O, lane, kc);
#pragma unroll
      for (int g = 0; g < 4; ++g)
#pragma unroll
        for (int t = 0; t < 2; ++t) {
          v16bf b = load_b_g(wk, G, lane, kc, g * U + u0 + 16 * t);
          acc[g][t][0] = __builtin_amdgcn_wmma_f32_16x16x32_bf16(
              false, a0, false, b, (short)0, acc[g][t][0], false, false);
          acc[g][t][1] = __builtin_amdgcn_wmma_f32_16x16x32_bf16(
              false, a1, false, b, (short)0, acc[g][t][1], false, false);
        }
    }
    // h part: K = 256
#pragma unroll
    for (int kc = 0; kc < U; kc += 32) {
      v16bf a0 = load_a_lds(sh_h,          U, lane, kc);
      v16bf a1 = load_a_lds(sh_h + 16 * U, U, lane, kc);
#pragma unroll
      for (int g = 0; g < 4; ++g)
#pragma unroll
        for (int t = 0; t < 2; ++t) {
          v16bf b = load_b_g(wr, G, lane, kc, g * U + u0 + 16 * t);
          acc[g][t][0] = __builtin_amdgcn_wmma_f32_16x16x32_bf16(
              false, a0, false, b, (short)0, acc[g][t][0], false, false);
          acc[g][t][1] = __builtin_amdgcn_wmma_f32_16x16x32_bf16(
              false, a1, false, b, (short)0, acc[g][t][1], false, false);
        }
    }
    __syncthreads();  // all waves done reading sh_h/sh_x before h rewrite

    // ============ Phase 2: elementwise cell update, write h to LDS =======
#pragma unroll
    for (int mt = 0; mt < MT; ++mt)
#pragma unroll
      for (int t = 0; t < 2; ++t)
#pragma unroll
        for (int r = 0; r < 8; ++r) {
          float zi = acc[0][t][mt][r];
          float zf = acc[1][t][mt][r];
          float zg = acc[2][t][mt][r];
          float zo = acc[3][t][mt][r];
          float cn = fast_sigmoid(zf) * cfr[mt][t][r] +
                     fast_sigmoid(zi) * fast_tanh(zg);
          cfr[mt][t][r] = cn;
          float hn = fast_sigmoid(zo) * fast_tanh(cn);
          sh_h[(mt * 16 + r + 8 * half) * U + u0 + 16 * t + col] = f2bf(hn);
        }
    __syncthreads();  // h_new visible to all waves

    // ============ Phase 3: dense  y = relu(h @ Wd + b) ============
    v8f accd[MT];
#pragma unroll
    for (int mt = 0; mt < MT; ++mt)
#pragma unroll
      for (int r = 0; r < 8; ++r) accd[mt][r] = db;
#pragma unroll
    for (int kc = 0; kc < U; kc += 32) {
      v16bf a0 = load_a_lds(sh_h,          U, lane, kc);
      v16bf a1 = load_a_lds(sh_h + 16 * U, U, lane, kc);
      v16bf b  = load_b_g(wd, O, lane, kc, o0);
      accd[0] = __builtin_amdgcn_wmma_f32_16x16x32_bf16(
          false, a0, false, b, (short)0, accd[0], false, false);
      accd[1] = __builtin_amdgcn_wmma_f32_16x16x32_bf16(
          false, a1, false, b, (short)0, accd[1], false, false);
    }
#pragma unroll
    for (int mt = 0; mt < MT; ++mt)
#pragma unroll
      for (int r = 0; r < 8; ++r) {
        float y = fmaxf(accd[mt][r], 0.0f);
        int row = mt * 16 + r + 8 * half;
        out[(size_t)(b0 + row) * (S * O) + (size_t)s * O + o0 + col] = y;
        sh_x[row * O + o0 + col] = f2bf(y);  // y becomes next step's x
      }
    __syncthreads();  // x_new visible for next iteration
  }
}

// ---------------------------------------------------------------------------
// Host launcher
// ---------------------------------------------------------------------------
extern "C" void kernel_launch(void* const* d_in, const int* in_sizes, int n_in,
                              void* d_out, int out_size, void* d_ws, size_t ws_size,
                              hipStream_t stream) {
  const float* last_input = (const float*)d_in[0];  // [B,O]
  const float* h0         = (const float*)d_in[1];  // [B,U]
  const float* c0         = (const float*)d_in[2];  // [B,U]
  const float* kernel     = (const float*)d_in[3];  // [O,4U]
  const float* rec_kernel = (const float*)d_in[4];  // [U,4U]
  const float* bias       = (const float*)d_in[5];  // [4U]
  const float* dense_w    = (const float*)d_in[6];  // [U,O]
  const float* dense_b    = (const float*)d_in[7];  // [O]
  float* out = (float*)d_out;                       // [B,S,O]

  const int n_wk = O * G;   // 131072
  const int n_wr = U * G;   // 262144
  const int n_wd = U * O;   // 32768

  unsigned short* wk = (unsigned short*)d_ws;
  unsigned short* wr = wk + n_wk;
  unsigned short* wd = wr + n_wr;

  // fp32 -> bf16 weight prep (cheap, deterministic, runs every call)
  cvt_bf16_kernel<<<(n_wk + 255) / 256, 256, 0, stream>>>(kernel,     wk, n_wk);
  cvt_bf16_kernel<<<(n_wr + 255) / 256, 256, 0, stream>>>(rec_kernel, wr, n_wr);
  cvt_bf16_kernel<<<(n_wd + 255) / 256, 256, 0, stream>>>(dense_w,    wd, n_wd);

  // persistent decoder: 128 batch tiles x 8 waves
  lstm_decoder_kernel<<<dim3(Bsz / ROWS), dim3(256), 0, stream>>>(
      last_input, h0, c0, bias, dense_b, wk, wr, wd, out);
}
